// STULayer_28853590294871
// MI455X (gfx1250) — matrix operand order repaired and checked by reading, not compiled, not run on hardware
//
#include <hip/hip_runtime.h>
#include <stdint.h>

// ---------------------------------------------------------------------------
// STU layer for MI455X (gfx1250, wave32, WMMA).
//   out[b,t,d] = 2*tanh(gate) * sum_{m>=0} phi[2m,? ] ... via k-factorization:
//   out[b,t,d] = 2*tanh(g) * sum_k Mf[k,d] * (E_k (*) xproj[:,d])[t],  E_k[m]=phi[2m,k]
// Parity split: even lags only -> two independent length-1024 streams.
// Phase 1: xproj = x @ Mi as bf16 WMMA GEMM, stored parity-deinterleaved,
//          channel-major: xq[b][par][d][u]  (L2-resident, 32MB).
// Phase 2: 24 shared-filter convolutions as Toeplitz-block WMMA
//          (A = pre-swizzled filter-Toeplitz fragments, B = xq tile),
//          fused per-lane contraction with Mf and gate scaling.
// ---------------------------------------------------------------------------

typedef __bf16 bf16;
typedef __attribute__((ext_vector_type(16))) __bf16   v16bf;
typedef __attribute__((ext_vector_type(8)))  float    v8f;
typedef __attribute__((ext_vector_type(4)))  uint32_t u32x4;
typedef __attribute__((ext_vector_type(4)))  __bf16   bf16x4;

union FragA { u32x4 q[2]; v16bf v; };

#define DM   1024      // d_model
#define TT   2048      // seq len
#define BB   8         // batch
#define NK   24        // num filters
#define UU   1024      // per-parity length
#define TA_HW_PER_D16 (NK * 512)   // halfwords per Delta16 chunk (24 frags * 1KB)

// ---------------- prep: x fp32 -> bf16 (same [b][t][e] layout) --------------
__global__ __launch_bounds__(256) void k_prep_xb(const float* __restrict__ x,
                                                 bf16* __restrict__ xb) {
  int idx = blockIdx.x * 256 + threadIdx.x;          // 4 elems/thread
  float4 v = ((const float4*)x)[idx];
  bf16x4 o;
  o.x = (bf16)v.x; o.y = (bf16)v.y; o.z = (bf16)v.z; o.w = (bf16)v.w;
  ((bf16x4*)xb)[idx] = o;
}

// ---------------- prep: MiT[d][e] = bf16(Mi[e][d]) --------------------------
__global__ __launch_bounds__(256) void k_prep_miT(const float* __restrict__ Mi,
                                                  bf16* __restrict__ miT) {
  int idx = blockIdx.x * 256 + threadIdx.x;          // 1M threads
  int d = idx >> 10, e = idx & 1023;
  miT[idx] = (bf16)Mi[e * DM + d];
}

// -------- prep: pre-swizzled filter-Toeplitz WMMA A fragments ---------------
// TA[d16][k] is a 16x32 bf16 A-fragment: A[t][j] = E_k[16*d16 + t - j],
// E_k[m] = phi[2m][k] (even-lag taps), zero outside [0,1024) (=> causality).
// Stored in per-lane contiguous-K layout (same as B) as two 16B chunks per lane:
//   halfword flat = ((d16*24 + k)*512) + c*256 + lane*8 + hw   (c = K-run)
__global__ __launch_bounds__(256) void k_prep_tableA(const float* __restrict__ phi,
                                                     bf16* __restrict__ tableA) {
  int idx = blockIdx.x * 256 + threadIdx.x;          // 786432 halfwords
  int hw   = idx & 7;
  int lane = (idx >> 3) & 31;
  int c    = (idx >> 8) & 1;
  int kf   = (idx >> 9) % NK;
  int d16  = (idx >> 9) / NK;
  int h  = c * 8 + hw;
  int hi = lane >> 4;
  int M  = lane & 15;
  int K  = h + 16 * hi;                              // contiguous K: lanes0-15 K=0..15, lanes16-31 K=16..31
  int m  = 16 * d16 + M - K;                         // even-lag tap index
  float val = (m >= 0 && m < UU) ? phi[(2 * m) * NK + kf] : 0.0f;
  tableA[idx] = (bf16)val;
}

// ---------------- phase 1: xq = (x @ Mi), bf16 WMMA GEMM --------------------
// wave macro-tile: 64 d x 64 t (16 accumulators). A = MiT rows, B = xb rows.
// Epilogue writes parity-deinterleaved channel-major xq[b][par][d][u].
__global__ __launch_bounds__(256) void k_gemm(const bf16* __restrict__ xb,
                                              const bf16* __restrict__ miT,
                                              bf16* __restrict__ xq) {
  int wave = (blockIdx.x * 256 + threadIdx.x) >> 5;  // 4096 waves
  int lane = threadIdx.x & 31;
  int tg = wave & 31;            // 32 t-groups of 64
  int dg = (wave >> 5) & 15;     // 16 d-groups of 64
  int b  = wave >> 9;            // 8 batches
  int dbase = dg * 64, tbase = tg * 64;
  int lm = lane & 15, hi = lane >> 4;

  const v8f vzero = {0.f,0.f,0.f,0.f,0.f,0.f,0.f,0.f};
  v8f acc[4][4];
  for (int i = 0; i < 4; ++i)
    for (int j = 0; j < 4; ++j) acc[i][j] = vzero;

  for (int e = 0; e < DM; e += 32) {
    v16bf Bfr[4];
#pragma unroll
    for (int tt = 0; tt < 4; ++tt) {
      // B 32x16: lanes0-15 K=0..15 (N=lane), lanes16-31 K=16..31; contiguous in e
      const bf16* bp = xb + ((size_t)(b * TT + tbase + tt * 16 + lm) * DM + e + hi * 16);
      Bfr[tt] = *(const v16bf*)bp;
    }
#pragma unroll
    for (int dt = 0; dt < 4; ++dt) {
      // A 16x32: lanes0-15 K=0..15, lanes16-31 K=16..31 (contiguous, matches B)
      const bf16* ap = miT + ((size_t)(dbase + dt * 16 + lm) * DM + e + hi * 16);
      FragA A;
      A.q[0] = *(const u32x4*)(ap);
      A.q[1] = *(const u32x4*)(ap + 8);
#pragma unroll
      for (int tt = 0; tt < 4; ++tt)
        acc[dt][tt] = __builtin_amdgcn_wmma_f32_16x16x32_bf16(
            false, A.v, false, Bfr[tt], (short)0, acc[dt][tt], false, false);
    }
  }
  // C/D layout: VGPR v -> M = v + 8*hi (=d_local), N = lane&15 (=t_local)
#pragma unroll
  for (int dt = 0; dt < 4; ++dt)
#pragma unroll
    for (int tt = 0; tt < 4; ++tt)
#pragma unroll
      for (int v = 0; v < 8; ++v) {
        int d = dbase + dt * 16 + v + 8 * hi;
        int t = tbase + tt * 16 + lm;
        int par = t & 1, u = t >> 1;
        xq[((size_t)(b * 2 + par) * DM + d) * UU + u] = (bf16)acc[dt][tt][v];
      }
}

// ------- phase 2: Toeplitz-WMMA conv (24 filters) + k-contraction -----------
// block = 256 thr = 8 waves = 4 channel-tiles x 2 k-groups(12 filters each).
// Per lag-step: stage 24KB of A fragments to LDS (shared by 4 ch-tiles),
// each wave: 1 B-frag global load + 12 LDS A-frags + 12 wmma.
__global__ __launch_bounds__(256) void k_conv(const bf16* __restrict__ xq,
                                              const bf16* __restrict__ tableA,
                                              const float* __restrict__ Mf,
                                              const float* __restrict__ gate,
                                              float* __restrict__ out) {
  __shared__ bf16  ldsA[NK * 512];     // 24KB: [k][c][lane][8 hw]
  __shared__ float ldsR[4 * 8 * 32];   // 4KB partial-out exchange

  int tid  = threadIdx.x;
  int lane = tid & 31, wave = tid >> 5;
  int kg = wave & 1, ct = wave >> 1;
  int i  = blockIdx.x;                 // 0..63 output u-tile
  int cg = blockIdx.y;                 // 0..15 channel group (64 ch)
  int bp = blockIdx.z;                 // 0..15 = b*2 + parity
  int dbase = cg * 64 + ct * 16;
  int lm = lane & 15, hi = lane >> 4;

  const v8f vzero = {0.f,0.f,0.f,0.f,0.f,0.f,0.f,0.f};
  v8f acc[12];
#pragma unroll
  for (int k = 0; k < 12; ++k) acc[k] = vzero;

  const bf16* xrow = xq + (size_t)bp * DM * UU;
  int nsteps = (i >> 1) + 1;

  for (int s = 0; s < nsteps; ++s) {
    int d16 = i - 2 * s;               // Delta/16, >= 0
    __syncthreads();
    {
      const u32x4* src = (const u32x4*)(tableA + (size_t)d16 * TA_HW_PER_D16);
      u32x4* dst = (u32x4*)ldsA;       // 1536 u32x4 = 256 thr * 6
#pragma unroll
      for (int r = 0; r < 6; ++r) dst[tid + 256 * r] = src[tid + 256 * r];
    }
    if (d16 >= 2)                      // pull next step's fragments toward L0/L2
      __builtin_prefetch(tableA + (size_t)(d16 - 2) * TA_HW_PER_D16, 0, 1);
    __syncthreads();

    int sbase = 32 * s;
    // B 32x16: lane column = channel, 16 contiguous u values per lane
    v16bf Bfr = *(const v16bf*)(xrow + (size_t)(dbase + lm) * UU + sbase + hi * 16);
    const u32x4* la = (const u32x4*)ldsA;
#pragma unroll
    for (int k12 = 0; k12 < 12; ++k12) {
      int k = kg * 12 + k12;
      FragA A;                          // conflict-friendly 16B-per-lane layout
      A.q[0] = la[(k * 2 + 0) * 32 + lane];
      A.q[1] = la[(k * 2 + 1) * 32 + lane];
      acc[k12] = __builtin_amdgcn_wmma_f32_16x16x32_bf16(
          false, A.v, false, Bfr, (short)0, acc[k12], false, false);
    }
  }

  // epilogue: per-lane k-contraction with Mf, cross-kgroup LDS reduce, scale
  float scale = 2.0f * tanhf(gate[0]);
  int d = dbase + lm;                  // C layout: N = lane&15
  float po[8];
#pragma unroll
  for (int v = 0; v < 8; ++v) po[v] = 0.f;
#pragma unroll
  for (int k12 = 0; k12 < 12; ++k12) {
    float mf = Mf[(kg * 12 + k12) * DM + d];
#pragma unroll
    for (int v = 0; v < 8; ++v) po[v] += mf * acc[k12][v];
  }
  __syncthreads();
  if (kg == 1) {
#pragma unroll
    for (int v = 0; v < 8; ++v) ldsR[(ct * 8 + v) * 32 + lane] = po[v];
  }
  __syncthreads();
  if (kg == 0) {
    int b = bp >> 1, par = bp & 1;
#pragma unroll
    for (int v = 0; v < 8; ++v) {
      float r = (po[v] + ldsR[(ct * 8 + v) * 32 + lane]) * scale;
      int u = i * 16 + v + 8 * hi;     // C layout: M = v + 8*hi
      int t = 2 * u + par;
      out[((size_t)b * TT + t) * DM + d] = r;
    }
  }
}

// ---------------------------------------------------------------------------
extern "C" void kernel_launch(void* const* d_in, const int* in_sizes, int n_in,
                              void* d_out, int out_size, void* d_ws, size_t ws_size,
                              hipStream_t stream) {
  const float* x    = (const float*)d_in[0];   // [8,2048,1024]
  const float* Mi   = (const float*)d_in[1];   // [1024,1024]
  const float* Mf   = (const float*)d_in[2];   // [24,1024]
  const float* gate = (const float*)d_in[3];   // scalar
  const float* phi  = (const float*)d_in[4];   // [2048,24]
  float* out = (float*)d_out;

  char* ws = (char*)d_ws;
  bf16* xb     = (bf16*)(ws);                                   // 33.5 MB
  bf16* miT    = (bf16*)(ws + 33554432);                        //  2.1 MB
  bf16* xq     = (bf16*)(ws + 33554432 + 2097152);              // 33.5 MB
  bf16* tableA = (bf16*)(ws + 33554432 + 2097152 + 33554432);   //  1.6 MB

  k_prep_xb    <<<16384, 256, 0, stream>>>(x, xb);
  k_prep_miT   <<<4096,  256, 0, stream>>>(Mi, miT);
  k_prep_tableA<<<3072,  256, 0, stream>>>(phi, tableA);
  k_gemm       <<<512,   256, 0, stream>>>(xb, miT, xq);
  k_conv       <<<dim3(64, 16, 16), 256, 0, stream>>>(xq, tableA, Mf, gate, out);
}
